// PaliGemma2WithExpertModel_41893111005339
// MI455X (gfx1250) — matrix operand order repaired and checked by reading, not compiled, not run on hardware
//
#include <hip/hip_runtime.h>

// ---------------------------------------------------------------------------
// CDNA5 / gfx1250: wave32, WMMA 16x16x32 bf16 (f32 accumulate)
// ---------------------------------------------------------------------------

typedef __bf16 bf16_t;
typedef __attribute__((ext_vector_type(16))) __bf16 v16bf;
typedef __attribute__((ext_vector_type(8)))  __bf16 v8bf;
typedef __attribute__((ext_vector_type(8)))  float  v8f;

constexpr int kB  = 2;
constexpr int kLP = 1024;
constexpr int kLE = 1024;
constexpr int kL  = 2048;
constexpr int kH  = 8;
constexpr int kKV = 4;
constexpr int kD  = 256;
constexpr int kG  = kH / kKV;       // 2
constexpr int kDH = 2304;
constexpr int kDE = 1024;
constexpr int kDHALF = 128;
constexpr int kHD  = kH * kD;       // 2048
constexpr int kKVD = kKV * kD;      // 1024
constexpr float kScale = 0.0625f;   // 256^-0.5

// ---------------------------------------------------------------------------
// fp32 -> bf16 conversion
// ---------------------------------------------------------------------------
__global__ void cvt_f32_bf16(const float* __restrict__ in, bf16_t* __restrict__ out, long n) {
    long i = (long)blockIdx.x * blockDim.x + threadIdx.x;
    if (i < n) out[i] = (bf16_t)in[i];
}

// ---------------------------------------------------------------------------
// A-fragment assembly: p points at (row_base + k + koffA); one contiguous
// 32-byte run per lane (k = 16*half + s), matching the B-fragment order.
// ---------------------------------------------------------------------------
__device__ __forceinline__ v16bf make_a_frag(const bf16_t* p) {
    v8bf lo = *(const v8bf*)(p);
    v8bf hi = *(const v8bf*)(p + 8);
    v16bf a;
#pragma unroll
    for (int i = 0; i < 8; ++i) { a[i] = lo[i]; a[i + 8] = hi[i]; }
    return a;
}

__device__ __forceinline__ v8f wmma_bf16(v16bf a, v16bf b, v8f c) {
    return __builtin_amdgcn_wmma_f32_16x16x32_bf16(false, a, false, b, (short)0, c, false, false);
}

// ---------------------------------------------------------------------------
// GEMM: Y[b][m][n] = sum_k X[b][m][k] * W[n][k]   (W shared across batch)
// X row-major (ldx), W row-major NxK, Y row-major (ldy).
// One wave computes a 32x64 tile: 2 A-frags x 4 B-frags -> 8 WMMAs per
// K-step against 8 VMEM loads (1:1 load:WMMA issue ratio).
// ---------------------------------------------------------------------------
template <bool F32OUT>
__global__ void gemm_bf16_kernel(const bf16_t* __restrict__ X, long xstride_b, int ldx,
                                 const bf16_t* __restrict__ W,
                                 void* __restrict__ Y, long ystride_b, int ldy,
                                 int M, int N, int K) {
    int lane = threadIdx.x & 31;
    int wv   = threadIdx.x >> 5;
    long wid = (long)blockIdx.x * (blockDim.x >> 5) + wv;
    int mtiles = M >> 5;                 // 32 rows per wave tile
    int ntiles = N >> 6;                 // 64 cols per wave tile
    if (wid >= (long)mtiles * ntiles) return;
    int mi = (int)(wid / ntiles);
    int ni = (int)(wid - (long)mi * ntiles);
    int bz = blockIdx.z;

    const bf16_t* Xb = X + (long)bz * xstride_b;
    int m0 = mi << 5, n0 = ni << 6;

    int koffA = (lane & 16) ? 16 : 0;
    int koffB = (lane & 16) ? 16 : 0;
    const bf16_t* xrow0 = Xb + (long)(m0 + (lane & 15)) * ldx + koffA;
    const bf16_t* xrow1 = xrow0 + (long)16 * ldx;
    const bf16_t* wrow0 = W + (long)(n0 + (lane & 15)) * K + koffB;

    v8f c[8];
#pragma unroll
    for (int t = 0; t < 8; ++t) c[t] = v8f{};

    for (int k = 0; k < K; k += 32) {
        v16bf a0 = make_a_frag(xrow0 + k);
        v16bf a1 = make_a_frag(xrow1 + k);
        v16bf b0 = *(const v16bf*)(wrow0 + k);
        v16bf b1 = *(const v16bf*)(wrow0 + (long)16 * K + k);
        v16bf b2 = *(const v16bf*)(wrow0 + (long)32 * K + k);
        v16bf b3 = *(const v16bf*)(wrow0 + (long)48 * K + k);
        c[0] = wmma_bf16(a0, b0, c[0]);
        c[1] = wmma_bf16(a1, b0, c[1]);
        c[2] = wmma_bf16(a0, b1, c[2]);
        c[3] = wmma_bf16(a1, b1, c[3]);
        c[4] = wmma_bf16(a0, b2, c[4]);
        c[5] = wmma_bf16(a1, b2, c[5]);
        c[6] = wmma_bf16(a0, b3, c[6]);
        c[7] = wmma_bf16(a1, b3, c[7]);
    }

    int mrow = m0 + ((lane & 16) ? 8 : 0);
    int nc   = n0 + (lane & 15);
    if (F32OUT) {
        float* Yb = (float*)Y + (long)bz * ystride_b;
#pragma unroll
        for (int r = 0; r < 8; ++r) {
            float* p0 = Yb + (long)(mrow + r) * ldy + nc;        // rows m0..m0+15
            float* p1 = Yb + (long)(mrow + 16 + r) * ldy + nc;   // rows m0+16..m0+31
            p0[0] = c[0][r]; p0[16] = c[2][r]; p0[32] = c[4][r]; p0[48] = c[6][r];
            p1[0] = c[1][r]; p1[16] = c[3][r]; p1[32] = c[5][r]; p1[48] = c[7][r];
        }
    } else {
        bf16_t* Yb = (bf16_t*)Y + (long)bz * ystride_b;
#pragma unroll
        for (int r = 0; r < 8; ++r) {
            bf16_t* p0 = Yb + (long)(mrow + r) * ldy + nc;
            bf16_t* p1 = Yb + (long)(mrow + 16 + r) * ldy + nc;
            p0[0]  = (bf16_t)c[0][r];
            p0[16] = (bf16_t)c[2][r];
            p0[32] = (bf16_t)c[4][r];
            p0[48] = (bf16_t)c[6][r];
            p1[0]  = (bf16_t)c[1][r];
            p1[16] = (bf16_t)c[3][r];
            p1[32] = (bf16_t)c[5][r];
            p1[48] = (bf16_t)c[7][r];
        }
    }
}

// ---------------------------------------------------------------------------
// RoPE in place on [B, L, nh, 256] bf16 (pairs d / d+128), fp32 math.
// ---------------------------------------------------------------------------
__global__ void rope_kernel(bf16_t* __restrict__ x, const float* __restrict__ cosb,
                            const float* __restrict__ sinb, int nh, long total) {
    long i = (long)blockIdx.x * blockDim.x + threadIdx.x;
    if (i >= total) return;
    int  d = (int)(i & (kDHALF - 1));
    long t = i >> 7;                 // (b*L + pos)*nh + h
    int  pos = (int)((t / nh) % kL);
    long base = t * kD;
    float x1 = (float)x[base + d];
    float x2 = (float)x[base + kDHALF + d];
    float c = cosb[(long)pos * kDHALF + d];
    float s = sinb[(long)pos * kDHALF + d];
    x[base + d]          = (bf16_t)(x1 * c - x2 * s);
    x[base + kDHALF + d] = (bf16_t)(x2 * c + x1 * s);
}

// ---------------------------------------------------------------------------
// V transpose: v [B, L, KV, D] -> vt [B, KV, D, L]  (so P*V B-frags are
// contiguous 32-byte loads)
// ---------------------------------------------------------------------------
__global__ void transpose_v_kernel(const bf16_t* __restrict__ v, bf16_t* __restrict__ vt, long total) {
    long i = (long)blockIdx.x * blockDim.x + threadIdx.x;
    if (i >= total) return;
    int  l = (int)(i % kL);  long t1 = i / kL;
    int  d = (int)(t1 % kD); long t2 = t1 / kD;
    int  kv = (int)(t2 % kKV);
    int  b  = (int)(t2 / kKV);
    vt[i] = v[(((long)b * kL + l) * kKV + kv) * kD + d];
}

// ---------------------------------------------------------------------------
// Flash attention: one wave owns a 16-query tile for one (b, h).
// Online softmax over 32-key chunks; S = Q K^T via WMMA, P staged through
// wave-private LDS (C-layout -> A-layout), att += P V via WMMA.
// ---------------------------------------------------------------------------
__global__ __launch_bounds__(128) void attention_kernel(
    const bf16_t* __restrict__ qbuf, const bf16_t* __restrict__ kbuf,
    const bf16_t* __restrict__ vt, const float* __restrict__ mask,
    bf16_t* __restrict__ attb) {
    __shared__ __align__(64) bf16_t pbuf[4][16 * 32];

    int lane = threadIdx.x & 31;
    int wv   = threadIdx.x >> 5;
    int qt   = blockIdx.x * 4 + wv;     // 0 .. L/16-1
    int h    = blockIdx.y;
    int b    = blockIdx.z;
    int kv   = h / kG;
    int qbase = qt << 4;

    int koffA = (lane & 16) ? 16 : 0;
    int koffB = (lane & 16) ? 16 : 0;
    int mhalf = (lane & 16) ? 8 : 0;

    // Preload Q A-fragments for all 8 K-steps of D=256
    const bf16_t* qrow = qbuf + ((long)(b * kL + qbase + (lane & 15)) * kH + h) * kD + koffA;
    v16bf qf[8];
#pragma unroll
    for (int ks = 0; ks < 8; ++ks) qf[ks] = make_a_frag(qrow + ks * 32);

    const bf16_t* kbase_p = kbuf + (long)b * kL * kKVD + (long)kv * kD + koffB;
    const bf16_t* vbase_p = vt + ((long)(b * kKV + kv) * kD) * (long)kL + koffB;
    const float*  mbase_p = mask + (long)b * kL * kL;

    v8f acc[16];
#pragma unroll
    for (int t = 0; t < 16; ++t) acc[t] = v8f{};
    float Mrow[8], Lrow[8];
#pragma unroll
    for (int r = 0; r < 8; ++r) { Mrow[r] = -1e30f; Lrow[r] = 0.0f; }

    for (int kc = 0; kc < kL; kc += 32) {
        // ---- scores: 16 x 32 chunk via 2 WMMA accumulators ----
        v8f s0 = {}, s1 = {};
        const bf16_t* k0 = kbase_p + (long)(kc + (lane & 15)) * kKVD;
        const bf16_t* k1 = k0 + (long)16 * kKVD;
#pragma unroll
        for (int ds = 0; ds < 8; ++ds) {
            v16bf bk0 = *(const v16bf*)(k0 + ds * 32);
            v16bf bk1 = *(const v16bf*)(k1 + ds * 32);
            s0 = wmma_bf16(qf[ds], bk0, s0);
            s1 = wmma_bf16(qf[ds], bk1, s1);
        }

        int col0 = kc + (lane & 15);
        // ---- online softmax (row-wise across the 16-lane half) ----
#pragma unroll
        for (int r = 0; r < 8; ++r) {
            int qr = qbase + r + mhalf;
            const float* mr = mbase_p + (long)qr * kL;
            float v0 = s0[r] * kScale + mr[col0];
            float v1 = s1[r] * kScale + mr[col0 + 16];
            float mx = fmaxf(v0, v1);
#pragma unroll
            for (int off = 8; off >= 1; off >>= 1) mx = fmaxf(mx, __shfl_xor(mx, off, 32));
            float mnew  = fmaxf(Mrow[r], mx);
            float alpha = __expf(Mrow[r] - mnew);
            float p0 = __expf(v0 - mnew);
            float p1 = __expf(v1 - mnew);
            float rs = p0 + p1;
#pragma unroll
            for (int off = 8; off >= 1; off >>= 1) rs += __shfl_xor(rs, off, 32);
            Lrow[r] = Lrow[r] * alpha + rs;
            Mrow[r] = mnew;
#pragma unroll
            for (int t = 0; t < 16; ++t) acc[t][r] *= alpha;
            pbuf[wv][(r + mhalf) * 32 + (lane & 15)]      = (bf16_t)p0;
            pbuf[wv][(r + mhalf) * 32 + 16 + (lane & 15)] = (bf16_t)p1;
        }

        // ---- P (C-layout) -> A-fragment via wave-private LDS ----
        v16bf pa = make_a_frag(&pbuf[wv][(lane & 15) * 32 + koffA]);

        // ---- att += P (16x32) x V (32x16 per d-tile) ----
        const bf16_t* vb = vbase_p + kc;
#pragma unroll
        for (int t = 0; t < 16; ++t) {
            int dcol = t * 16 + (lane & 15);
            v16bf bv = *(const v16bf*)(vb + (long)dcol * kL);
            acc[t] = wmma_bf16(pa, bv, acc[t]);
        }
    }

    // ---- normalize + store att [B, L, H*D] as bf16 ----
#pragma unroll
    for (int r = 0; r < 8; ++r) {
        float inv = 1.0f / Lrow[r];
        long row = (long)b * kL + qbase + r + mhalf;
        bf16_t* orow = attb + row * kHD + h * kD + (lane & 15);
#pragma unroll
        for (int t = 0; t < 16; ++t) orow[t * 16] = (bf16_t)(acc[t][r] * inv);
    }
}

// ---------------------------------------------------------------------------
// Host side
// ---------------------------------------------------------------------------
static void gemm_launch(bool f32out,
                        const bf16_t* X, long xsb, int ldx, const bf16_t* W,
                        void* Y, long ysb, int ldy,
                        int M, int N, int K, hipStream_t s) {
    long waves = (long)(M / 32) * (N / 64);
    dim3 grid((unsigned)((waves + 7) / 8), 1, kB);
    if (f32out)
        gemm_bf16_kernel<true><<<grid, 256, 0, s>>>(X, xsb, ldx, W, Y, ysb, ldy, M, N, K);
    else
        gemm_bf16_kernel<false><<<grid, 256, 0, s>>>(X, xsb, ldx, W, Y, ysb, ldy, M, N, K);
}

extern "C" void kernel_launch(void* const* d_in, const int* in_sizes, int n_in,
                              void* d_out, int out_size, void* d_ws, size_t ws_size,
                              hipStream_t stream) {
    (void)in_sizes; (void)n_in; (void)out_size; (void)ws_size;
    const float* pg   = (const float*)d_in[0];
    const float* expe = (const float*)d_in[1];
    const float* cosb = (const float*)d_in[2];
    const float* sinb = (const float*)d_in[3];
    const float* mask = (const float*)d_in[4];
    const float* wq0  = (const float*)d_in[5];
    const float* wk0  = (const float*)d_in[6];
    const float* wv0  = (const float*)d_in[7];
    const float* wo0  = (const float*)d_in[8];
    const float* wq1  = (const float*)d_in[9];
    const float* wk1  = (const float*)d_in[10];
    const float* wv1  = (const float*)d_in[11];
    const float* wo1  = (const float*)d_in[12];

    char* ws = (char*)d_ws;
    size_t off = 0;
    auto take = [&](long elems) -> bf16_t* {
        bf16_t* p = (bf16_t*)(ws + off);
        off += ((size_t)elems * 2 + 255) & ~(size_t)255;
        return p;
    };

    bf16_t* xpg  = take((long)kB * kLP * kDH);
    bf16_t* xexp = take((long)kB * kLE * kDE);
    bf16_t* wq0b = take((long)kHD * kDH);
    bf16_t* wk0b = take((long)kKVD * kDH);
    bf16_t* wv0b = take((long)kKVD * kDH);
    bf16_t* wo0b = take((long)kDH * kHD);
    bf16_t* wq1b = take((long)kHD * kDE);
    bf16_t* wk1b = take((long)kKVD * kDE);
    bf16_t* wv1b = take((long)kKVD * kDE);
    bf16_t* wo1b = take((long)kDE * kHD);
    bf16_t* qbuf = take((long)kB * kL * kHD);
    bf16_t* kbuf = take((long)kB * kL * kKVD);
    bf16_t* vraw = take((long)kB * kL * kKVD);
    bf16_t* vtb  = take((long)kB * kL * kKVD);
    bf16_t* attb = take((long)kB * kL * kHD);

    auto cvt = [&](const float* src, bf16_t* dst, long n) {
        cvt_f32_bf16<<<dim3((unsigned)((n + 255) / 256)), dim3(256), 0, stream>>>(src, dst, n);
    };
    cvt(pg,   xpg,  (long)kB * kLP * kDH);
    cvt(expe, xexp, (long)kB * kLE * kDE);
    cvt(wq0, wq0b, (long)kHD * kDH);
    cvt(wk0, wk0b, (long)kKVD * kDH);
    cvt(wv0, wv0b, (long)kKVD * kDH);
    cvt(wo0, wo0b, (long)kDH * kHD);
    cvt(wq1, wq1b, (long)kHD * kDE);
    cvt(wk1, wk1b, (long)kKVD * kDE);
    cvt(wv1, wv1b, (long)kKVD * kDE);
    cvt(wo1, wo1b, (long)kDE * kHD);

    // QKV projections (bf16 out)
    gemm_launch(false, xpg,  (long)kLP * kDH, kDH, wq0b, qbuf,                 (long)kL * kHD,  kHD,  kLP, kHD,  kDH, stream);
    gemm_launch(false, xexp, (long)kLE * kDE, kDE, wq1b, qbuf + (long)kLP * kHD, (long)kL * kHD, kHD, kLE, kHD,  kDE, stream);
    gemm_launch(false, xpg,  (long)kLP * kDH, kDH, wk0b, kbuf,                 (long)kL * kKVD, kKVD, kLP, kKVD, kDH, stream);
    gemm_launch(false, xexp, (long)kLE * kDE, kDE, wk1b, kbuf + (long)kLP * kKVD, (long)kL * kKVD, kKVD, kLE, kKVD, kDE, stream);
    gemm_launch(false, xpg,  (long)kLP * kDH, kDH, wv0b, vraw,                 (long)kL * kKVD, kKVD, kLP, kKVD, kDH, stream);
    gemm_launch(false, xexp, (long)kLE * kDE, kDE, wv1b, vraw + (long)kLP * kKVD, (long)kL * kKVD, kKVD, kLE, kKVD, kDE, stream);

    // RoPE on Q and K
    {
        long nq = (long)kB * kL * kH * kDHALF;
        rope_kernel<<<dim3((unsigned)((nq + 255) / 256)), dim3(256), 0, stream>>>(qbuf, cosb, sinb, kH, nq);
        long nk = (long)kB * kL * kKV * kDHALF;
        rope_kernel<<<dim3((unsigned)((nk + 255) / 256)), dim3(256), 0, stream>>>(kbuf, cosb, sinb, kKV, nk);
    }

    // V -> d-major for WMMA B-fragments
    {
        long nv = (long)kB * kKV * kD * kL;
        transpose_v_kernel<<<dim3((unsigned)((nv + 255) / 256)), dim3(256), 0, stream>>>(vraw, vtb, nv);
    }

    // Flash attention
    attention_kernel<<<dim3(kL / 16 / 4, kH, kB), dim3(128), 0, stream>>>(qbuf, kbuf, vtb, mask, attb);

    // Output projections (f32 out, concatenated: out_pg then out_exp)
    float* out_pg  = (float*)d_out;
    float* out_exp = (float*)d_out + (long)kB * kLP * kDH;
    gemm_launch(true, attb,                  (long)kL * kHD, kHD, wo0b, out_pg,  (long)kLP * kDH, kDH, kLP, kDH, kHD, stream);
    gemm_launch(true, attb + (long)kLP * kHD, (long)kL * kHD, kHD, wo1b, out_exp, (long)kLE * kDE, kDE, kLE, kDE, kHD, stream);
}
